// AdaLNTransformerBlock_70300024701813
// MI455X (gfx1250) — matrix operand order repaired and checked by reading, not compiled, not run on hardware
//
#include <hip/hip_runtime.h>

typedef __attribute__((ext_vector_type(16))) __bf16 v16bf;
typedef __attribute__((ext_vector_type(8)))  float  v8f;

#define B_   4
#define S_   1024
#define H_   1024
#define NH_  16
#define DH_  64
#define NTOK (B_ * S_)

__device__ __forceinline__ unsigned short f2bf(float f) {
  unsigned u = __float_as_uint(f);
  u = u + 0x7FFFu + ((u >> 16) & 1u);   // round-to-nearest-even
  return (unsigned short)(u >> 16);
}

union FragU {
  v16bf v;
  uint4 q[2];
  unsigned short us[16];
};

// ---------------------------------------------------------------------------
// Weight convert + transpose: W[K][O] f32 -> Wt[O][K] bf16
// ---------------------------------------------------------------------------
__global__ void convT(const float* __restrict__ W, unsigned short* __restrict__ Wt,
                      int K, int O) {
  int n = K * O;
  for (int j = blockIdx.x * blockDim.x + threadIdx.x; j < n;
       j += gridDim.x * blockDim.x) {
    int o = j / K, k = j - o * K;
    Wt[j] = f2bf(W[(size_t)k * O + o]);
  }
}

// ---------------------------------------------------------------------------
// silu(t + h) -> bf16
// ---------------------------------------------------------------------------
__global__ void silu_add(const float* __restrict__ t, const float* __restrict__ h,
                         unsigned short* __restrict__ out, int n) {
  for (int i = blockIdx.x * blockDim.x + threadIdx.x; i < n;
       i += gridDim.x * blockDim.x) {
    float s = t[i] + h[i];
    out[i] = f2bf(s / (1.f + __expf(-s)));
  }
}

// ---------------------------------------------------------------------------
// GEMM: C[N][Hout] = A[N][K](bf16) @ Wt[Hout][K](bf16)^T + bias
// Register-blocked: one wave computes a 32x64 tile (2x4 accumulators),
// so 12 b128 loads feed 8 v_wmma_f32_16x16x32_bf16 per K-step, and the
// 8 independent accumulators pipeline through the matrix unit.
// Block = 256 threads = 8 waves (2x4) = 64x256 output tile.
// grid = (Hout/256, N/64). Optional relu / f32-residual / f32 out / bf16 out;
// obT=1 stores the bf16 output transposed ([Hout][N]) for attention V.
// ---------------------------------------------------------------------------
__global__ __launch_bounds__(256) void gemm_bf16(
    const unsigned short* __restrict__ A, const unsigned short* __restrict__ Wt,
    const float* __restrict__ bias, const float* __restrict__ residual,
    float* __restrict__ outF, unsigned short* __restrict__ outB,
    int N, int K, int Hout, int relu, int obT) {
  int lane = threadIdx.x & 31;
  int wave = threadIdx.x >> 5;
  int r = lane & 15, half = lane >> 4;
  int mt = wave >> 2, nt = wave & 3;
  int row0 = blockIdx.y * 64 + mt * 32;
  int col0 = blockIdx.x * 256 + nt * 64;

  const unsigned short* aBase0 = A + (size_t)(row0 + r) * K;
  const unsigned short* aBase1 = A + (size_t)(row0 + 16 + r) * K;
  const unsigned short* bBase[4];
#pragma unroll
  for (int ci = 0; ci < 4; ++ci)
    bBase[ci] = Wt + (size_t)(col0 + ci * 16 + r) * K;

  v8f acc[8] = {};
  for (int k0 = 0; k0 < K; k0 += 32) {
    FragU a0, a1, bf[4];
    // A-frag: lane half selects K blocks {0..7,16..23} vs {8..15,24..31}
    a0.q[0] = *(const uint4*)(aBase0 + k0 + 8 * half);
    a0.q[1] = *(const uint4*)(aBase0 + k0 + 16 + 8 * half);
    a1.q[0] = *(const uint4*)(aBase1 + k0 + 8 * half);
    a1.q[1] = *(const uint4*)(aBase1 + k0 + 16 + 8 * half);
    // B-frag: lane half selects K blocks {0..7,16..23} vs {8..15,24..31}
#pragma unroll
    for (int ci = 0; ci < 4; ++ci) {
      bf[ci].q[0] = *(const uint4*)(bBase[ci] + k0 + 8 * half);
      bf[ci].q[1] = *(const uint4*)(bBase[ci] + k0 + 16 + 8 * half);
    }
    __builtin_prefetch((const void*)(aBase0 + k0 + 512), 0, 1);
    __builtin_prefetch((const void*)(aBase1 + k0 + 512), 0, 1);
#pragma unroll
    for (int ci = 0; ci < 4; ++ci) {
      acc[ci] = __builtin_amdgcn_wmma_f32_16x16x32_bf16(
          false, a0.v, false, bf[ci].v, (short)0, acc[ci], false, false);
      acc[4 + ci] = __builtin_amdgcn_wmma_f32_16x16x32_bf16(
          false, a1.v, false, bf[ci].v, (short)0, acc[4 + ci], false, false);
    }
  }
#pragma unroll
  for (int mi = 0; mi < 2; ++mi) {
#pragma unroll
    for (int ci = 0; ci < 4; ++ci) {
      int col = col0 + ci * 16 + r;
      float bv = bias[col];
      v8f acv = acc[mi * 4 + ci];
#pragma unroll
      for (int i = 0; i < 8; ++i) {
        int row = row0 + mi * 16 + i + 8 * half;
        float v = acv[i] + bv;
        if (relu) v = fmaxf(v, 0.f);
        size_t idx = (size_t)row * Hout + col;
        if (residual) v += residual[idx];
        if (outF) outF[idx] = v;
        if (outB) {
          size_t bidx = obT ? ((size_t)col * N + row) : idx;
          outB[bidx] = f2bf(v);
        }
      }
    }
  }
}

// ---------------------------------------------------------------------------
// Flash attention: one wave per (b, head, 16-query tile).
// Q,K bf16 [NTOK][H_]; V is TRANSPOSED bf16 [H_][NTOK] so that P*V
// B-fragments are two contiguous b128 loads. O bf16 [NTOK][H_].
// rowmask: per-query-row int mask (0 -> -1e9 bias, matching the reference).
// ---------------------------------------------------------------------------
__global__ __launch_bounds__(256) void attn_wmma(
    const unsigned short* __restrict__ Q, const unsigned short* __restrict__ Kb,
    const unsigned short* __restrict__ Vt, const int* __restrict__ rowmask,
    unsigned short* __restrict__ O) {
  __shared__ __align__(16) unsigned short pS[8 * 16 * 32];
  int lane = threadIdx.x & 31;
  int wave = threadIdx.x >> 5;
  int r = lane & 15, half = lane >> 4;
  unsigned short* ps = pS + wave * 512;

  int wg = blockIdx.x * 8 + wave;
  int qt = wg & 63;              // S_/16 = 64 query tiles
  int head = (wg >> 6) & 15;
  int b = wg >> 10;
  int q0 = qt * 16;
  int hoff = head * DH_;
  size_t qrow = (size_t)(b * S_ + q0);

  FragU aq0, aq1;  // Q A-frags for k0 = 0 / 32 of DH
  {
    const unsigned short* p = Q + (qrow + r) * H_ + hoff;
    aq0.q[0] = *(const uint4*)(p + 16 * half);
    aq0.q[1] = *(const uint4*)(p + 16 * half + 8);
    aq1.q[0] = *(const uint4*)(p + 32 + 16 * half);
    aq1.q[1] = *(const uint4*)(p + 32 + 16 * half + 8);
  }
  float mb[8], m[8], lsum[8];
#pragma unroll
  for (int i = 0; i < 8; ++i) {
    int tq = q0 + i + 8 * half;
    mb[i] = (rowmask[b * S_ + tq] == 0) ? -1.0e9f : 0.f;
    m[i] = -3.0e38f;
    lsum[i] = 0.f;
  }
  v8f o0 = {}, o1 = {}, o2 = {}, o3 = {};
  // Per-lane V base: row dh = hoff + t4*16 + r of Vt, token kt + 16*half
  const unsigned short* vBase = Vt + (size_t)(hoff + r) * NTOK + b * S_;

  for (int kt = 0; kt < S_; kt += 32) {
    v8f s0 = {}, s1 = {};
    {
      FragU bk;
      const unsigned short* p = Kb + (size_t)(b * S_ + kt + r) * H_ + hoff;
      bk.q[0] = *(const uint4*)(p + 16 * half);
      bk.q[1] = *(const uint4*)(p + 16 * half + 8);
      s0 = __builtin_amdgcn_wmma_f32_16x16x32_bf16(false, aq0.v, false, bk.v,
                                                   (short)0, s0, false, false);
      bk.q[0] = *(const uint4*)(p + 32 + 16 * half);
      bk.q[1] = *(const uint4*)(p + 32 + 16 * half + 8);
      s0 = __builtin_amdgcn_wmma_f32_16x16x32_bf16(false, aq1.v, false, bk.v,
                                                   (short)0, s0, false, false);
      const unsigned short* p2 = p + 16 * H_;
      bk.q[0] = *(const uint4*)(p2 + 16 * half);
      bk.q[1] = *(const uint4*)(p2 + 16 * half + 8);
      s1 = __builtin_amdgcn_wmma_f32_16x16x32_bf16(false, aq0.v, false, bk.v,
                                                   (short)0, s1, false, false);
      bk.q[0] = *(const uint4*)(p2 + 32 + 16 * half);
      bk.q[1] = *(const uint4*)(p2 + 32 + 16 * half + 8);
      s1 = __builtin_amdgcn_wmma_f32_16x16x32_bf16(false, aq1.v, false, bk.v,
                                                   (short)0, s1, false, false);
    }
    float alpha[8];
#pragma unroll
    for (int i = 0; i < 8; ++i) {
      float x0 = s0[i] * 0.125f + mb[i];   // 1/sqrt(64)
      float x1 = s1[i] * 0.125f + mb[i];
      float tmax = fmaxf(x0, x1);
      for (int msk = 1; msk < 16; msk <<= 1)
        tmax = fmaxf(tmax, __shfl_xor(tmax, msk, 32));
      float mn = fmaxf(m[i], tmax);
      float a = __expf(m[i] - mn);
      float p0 = __expf(x0 - mn), p1 = __expf(x1 - mn);
      float rs = p0 + p1;
      for (int msk = 1; msk < 16; msk <<= 1) rs += __shfl_xor(rs, msk, 32);
      lsum[i] = lsum[i] * a + rs;
      m[i] = mn;
      alpha[i] = a;
      int row = i + 8 * half;                 // C-layout -> LDS row-major P
      ps[row * 32 + r] = f2bf(p0);
      ps[row * 32 + 16 + r] = f2bf(p1);
    }
#pragma unroll
    for (int i = 0; i < 8; ++i) {
      o0[i] *= alpha[i]; o1[i] *= alpha[i];
      o2[i] *= alpha[i]; o3[i] *= alpha[i];
    }
    asm volatile("s_wait_dscnt 0" ::: "memory");
    FragU pf;  // reload P in A-fragment layout
    pf.q[0] = *(const uint4*)(ps + r * 32 + 16 * half);
    pf.q[1] = *(const uint4*)(ps + r * 32 + 16 * half + 8);
    const unsigned short* vp = vBase + kt + 16 * half;
#pragma unroll
    for (int t4 = 0; t4 < 4; ++t4) {
      FragU vf;  // B-frag of Vt: contiguous tokens for column dh
      vf.q[0] = *(const uint4*)(vp + (size_t)(t4 * 16) * NTOK);
      vf.q[1] = *(const uint4*)(vp + (size_t)(t4 * 16) * NTOK + 8);
      v8f* op = (t4 == 0) ? &o0 : (t4 == 1) ? &o1 : (t4 == 2) ? &o2 : &o3;
      *op = __builtin_amdgcn_wmma_f32_16x16x32_bf16(false, pf.v, false, vf.v,
                                                    (short)0, *op, false, false);
    }
  }
#pragma unroll
  for (int i = 0; i < 8; ++i) {
    float inv = 1.f / lsum[i];
    size_t base = (qrow + i + 8 * half) * H_ + hoff + r;
    O[base + 0]  = f2bf(o0[i] * inv);
    O[base + 16] = f2bf(o1[i] * inv);
    O[base + 32] = f2bf(o2[i] * inv);
    O[base + 48] = f2bf(o3[i] * inv);
  }
}

// ---------------------------------------------------------------------------
// LayerNorm over H=1024, one block per token.
// y = ln(X + addres) [* g + b] [+ postres] -> optional f32 / bf16 outputs
// ---------------------------------------------------------------------------
__global__ __launch_bounds__(256) void ln_kernel(
    const float* __restrict__ X, const float* __restrict__ addres,
    const float* __restrict__ g, const float* __restrict__ bt,
    const float* __restrict__ postres, float* __restrict__ outF,
    unsigned short* __restrict__ outB) {
  __shared__ float s1[256], s2[256];
  int t = threadIdx.x;
  size_t base = (size_t)blockIdx.x * H_;
  float x[4];
  float sum = 0.f, sq = 0.f;
#pragma unroll
  for (int j = 0; j < 4; ++j) {
    int c = t + j * 256;
    float v = X[base + c];
    if (addres) v += addres[base + c];
    x[j] = v; sum += v; sq += v * v;
  }
  s1[t] = sum; s2[t] = sq; __syncthreads();
  for (int o = 128; o > 0; o >>= 1) {
    if (t < o) { s1[t] += s1[t + o]; s2[t] += s2[t + o]; }
    __syncthreads();
  }
  float mean = s1[0] * (1.f / H_);
  float var = s2[0] * (1.f / H_) - mean * mean;
  float rstd = rsqrtf(var + 1e-5f);
#pragma unroll
  for (int j = 0; j < 4; ++j) {
    int c = t + j * 256;
    float y = (x[j] - mean) * rstd;
    if (g) y = y * g[c] + bt[c];
    if (postres) y += postres[base + c];
    if (outF) outF[base + c] = y;
    if (outB) outB[base + c] = f2bf(y);
  }
}

// ---------------------------------------------------------------------------
// adaLN: ln(x) * (1 + scale) + shift -> bf16.  mod = [shift | scale] f32 [N][2H]
// ---------------------------------------------------------------------------
__global__ __launch_bounds__(256) void ln_mod_kernel(
    const float* __restrict__ X, const float* __restrict__ mod,
    unsigned short* __restrict__ outB) {
  __shared__ float s1[256], s2[256];
  int t = threadIdx.x;
  size_t base = (size_t)blockIdx.x * H_;
  size_t mbase = (size_t)blockIdx.x * (2 * H_);
  float x[4];
  float sum = 0.f, sq = 0.f;
#pragma unroll
  for (int j = 0; j < 4; ++j) {
    int c = t + j * 256;
    float v = X[base + c];
    x[j] = v; sum += v; sq += v * v;
  }
  s1[t] = sum; s2[t] = sq; __syncthreads();
  for (int o = 128; o > 0; o >>= 1) {
    if (t < o) { s1[t] += s1[t + o]; s2[t] += s2[t + o]; }
    __syncthreads();
  }
  float mean = s1[0] * (1.f / H_);
  float var = s2[0] * (1.f / H_) - mean * mean;
  float rstd = rsqrtf(var + 1e-5f);
#pragma unroll
  for (int j = 0; j < 4; ++j) {
    int c = t + j * 256;
    float y = (x[j] - mean) * rstd;
    y = y * (1.f + mod[mbase + H_ + c]) + mod[mbase + c];
    outB[base + c] = f2bf(y);
  }
}

// ---------------------------------------------------------------------------
extern "C" void kernel_launch(void* const* d_in, const int* in_sizes, int n_in,
                              void* d_out, int out_size, void* d_ws,
                              size_t ws_size, hipStream_t stream) {
  (void)in_sizes; (void)n_in; (void)out_size; (void)ws_size;
  const float* x = (const float*)d_in[0];
  const float* h = (const float*)d_in[1];
  const float* t = (const float*)d_in[2];
  const int* mask = (const int*)d_in[3];
  const int* extent = (const int*)d_in[4];
  const float* Wsq = (const float*)d_in[5];  const float* bsq = (const float*)d_in[6];
  const float* Wsk = (const float*)d_in[7];  const float* bsk = (const float*)d_in[8];
  const float* Wsv = (const float*)d_in[9];  const float* bsv = (const float*)d_in[10];
  const float* Wso = (const float*)d_in[11]; const float* bso = (const float*)d_in[12];
  const float* Wcq = (const float*)d_in[13]; const float* bcq = (const float*)d_in[14];
  const float* Wck = (const float*)d_in[15]; const float* bck = (const float*)d_in[16];
  const float* Wcv = (const float*)d_in[17]; const float* bcv = (const float*)d_in[18];
  const float* Wco = (const float*)d_in[19]; const float* bco = (const float*)d_in[20];
  const float* ada_w = (const float*)d_in[21]; const float* ada_b = (const float*)d_in[22];
  const float* fw1 = (const float*)d_in[23]; const float* fb1 = (const float*)d_in[24];
  const float* fw2 = (const float*)d_in[25]; const float* fb2 = (const float*)d_in[26];
  const float* ln2_g = (const float*)d_in[27]; const float* ln2_b = (const float*)d_in[28];
  const float* ln3_g = (const float*)d_in[29]; const float* ln3_b = (const float*)d_in[30];
  const float* lnh_g = (const float*)d_in[31]; const float* lnh_b = (const float*)d_in[32];
  const float* lnf_g = (const float*)d_in[33]; const float* lnf_b = (const float*)d_in[34];

  char* ws = (char*)d_ws;
  size_t off = 0;
  auto alloc = [&](size_t bytes) -> char* {
    char* p = ws + off;
    off += (bytes + 255) & ~(size_t)255;
    return p;
  };
  const size_t WB = (size_t)H_ * H_ * 2;         // 2 MiB bf16 weight
  unsigned short* wt_sq = (unsigned short*)alloc(WB);
  unsigned short* wt_sk = (unsigned short*)alloc(WB);
  unsigned short* wt_sv = (unsigned short*)alloc(WB);
  unsigned short* wt_so = (unsigned short*)alloc(WB);
  unsigned short* wt_cq = (unsigned short*)alloc(WB);
  unsigned short* wt_ck = (unsigned short*)alloc(WB);
  unsigned short* wt_cv = (unsigned short*)alloc(WB);
  unsigned short* wt_co = (unsigned short*)alloc(WB);
  unsigned short* wt_ada = (unsigned short*)alloc(WB * 2);
  unsigned short* wt_f1 = (unsigned short*)alloc(WB);
  unsigned short* wt_f2 = (unsigned short*)alloc(WB);

  const size_t ABF = (size_t)NTOK * H_ * 2;      // 8 MiB bf16 activation
  const size_t AF32 = (size_t)NTOK * H_ * 4;     // 16 MiB f32 activation
  unsigned short* sbf = (unsigned short*)alloc(ABF);   // silu buf, reused as ff1
  float* mod = (float*)alloc((size_t)NTOK * 2 * H_ * 4);
  float* R1 = mod;                               // mod dead after ln_mod
  float* R2 = (float*)((char*)mod + AF32);
  unsigned short* cbuf = (unsigned short*)alloc(ABF);  // hin / ln2 / ho bf16
  unsigned short* Qb = (unsigned short*)alloc(ABF);
  unsigned short* Kbf = (unsigned short*)alloc(ABF);
  unsigned short* Vtb = (unsigned short*)alloc(ABF);   // V transposed [H][NTOK]
  unsigned short* attnO = (unsigned short*)alloc(ABF);
  unsigned short* hnbf = (unsigned short*)alloc(ABF);
  float* hoF = (float*)alloc(AF32);
  float* Fbuf = (float*)alloc(AF32);

  // 1) weights -> bf16 transposed
  convT<<<2048, 256, 0, stream>>>(Wsq, wt_sq, H_, H_);
  convT<<<2048, 256, 0, stream>>>(Wsk, wt_sk, H_, H_);
  convT<<<2048, 256, 0, stream>>>(Wsv, wt_sv, H_, H_);
  convT<<<2048, 256, 0, stream>>>(Wso, wt_so, H_, H_);
  convT<<<2048, 256, 0, stream>>>(Wcq, wt_cq, H_, H_);
  convT<<<2048, 256, 0, stream>>>(Wck, wt_ck, H_, H_);
  convT<<<2048, 256, 0, stream>>>(Wcv, wt_cv, H_, H_);
  convT<<<2048, 256, 0, stream>>>(Wco, wt_co, H_, H_);
  convT<<<2048, 256, 0, stream>>>(ada_w, wt_ada, H_, 2 * H_);
  convT<<<2048, 256, 0, stream>>>(fw1, wt_f1, H_, H_);
  convT<<<2048, 256, 0, stream>>>(fw2, wt_f2, H_, H_);

  dim3 blk(256);
  dim3 gH(H_ / 256, NTOK / 64);        // Hout = 1024 GEMMs: (4, 64)
  dim3 g2H(2 * H_ / 256, NTOK / 64);   // Hout = 2048 GEMM:  (8, 64)

  // 2) adaLN modulation
  silu_add<<<4096, 256, 0, stream>>>(t, h, sbf, NTOK * H_);
  gemm_bf16<<<g2H, blk, 0, stream>>>(sbf, wt_ada, ada_b, nullptr, mod, nullptr,
                                     NTOK, H_, 2 * H_, 0, 0);
  ln_mod_kernel<<<NTOK, 256, 0, stream>>>(x, mod, cbuf);  // hidden_in bf16

  // 3) self attention
  gemm_bf16<<<gH, blk, 0, stream>>>(cbuf, wt_sq, bsq, nullptr, nullptr, Qb,
                                    NTOK, H_, H_, 0, 0);
  gemm_bf16<<<gH, blk, 0, stream>>>(cbuf, wt_sk, bsk, nullptr, nullptr, Kbf,
                                    NTOK, H_, H_, 0, 0);
  gemm_bf16<<<gH, blk, 0, stream>>>(cbuf, wt_sv, bsv, nullptr, nullptr, Vtb,
                                    NTOK, H_, H_, 0, 1);  // V transposed
  attn_wmma<<<512, 256, 0, stream>>>(Qb, Kbf, Vtb, extent, attnO);
  gemm_bf16<<<gH, blk, 0, stream>>>(attnO, wt_so, bso, x, R1, nullptr,
                                    NTOK, H_, H_, 0, 0);

  // 4) cross attention
  ln_kernel<<<NTOK, 256, 0, stream>>>(R1, nullptr, ln2_g, ln2_b, nullptr,
                                      nullptr, cbuf);
  ln_kernel<<<NTOK, 256, 0, stream>>>(h, nullptr, lnh_g, lnh_b, nullptr,
                                      nullptr, hnbf);
  gemm_bf16<<<gH, blk, 0, stream>>>(cbuf, wt_cq, bcq, nullptr, nullptr, Qb,
                                    NTOK, H_, H_, 0, 0);
  gemm_bf16<<<gH, blk, 0, stream>>>(hnbf, wt_ck, bck, nullptr, nullptr, Kbf,
                                    NTOK, H_, H_, 0, 0);
  gemm_bf16<<<gH, blk, 0, stream>>>(hnbf, wt_cv, bcv, nullptr, nullptr, Vtb,
                                    NTOK, H_, H_, 0, 1);  // V transposed
  attn_wmma<<<512, 256, 0, stream>>>(Qb, Kbf, Vtb, mask, attnO);
  gemm_bf16<<<gH, blk, 0, stream>>>(attnO, wt_co, bco, R1, R2, nullptr,
                                    NTOK, H_, H_, 0, 0);

  // 5) FFN
  ln_kernel<<<NTOK, 256, 0, stream>>>(R2, nullptr, ln3_g, ln3_b, nullptr,
                                      hoF, cbuf);                       // hidden_out
  gemm_bf16<<<gH, blk, 0, stream>>>(cbuf, wt_f1, fb1, nullptr, nullptr, sbf,
                                    NTOK, H_, H_, 1, 0);                // relu
  gemm_bf16<<<gH, blk, 0, stream>>>(sbf, wt_f2, fb2, nullptr, Fbuf, nullptr,
                                    NTOK, H_, H_, 0, 0);
  // 6) out = ln(ff2 + hidden_out) * g + b + hidden_mid
  ln_kernel<<<NTOK, 256, 0, stream>>>(Fbuf, hoF, lnf_g, lnf_b, R2,
                                      (float*)d_out, nullptr);
}
